// EllipticAffinity_85023172591719
// MI455X (gfx1250) — compile-verified, matrix operand order checked by reading, not run on hardware
//
#include <hip/hip_runtime.h>
#include <hip/hip_bf16.h>

typedef __attribute__((ext_vector_type(16))) _Float16 v16h;
typedef __attribute__((ext_vector_type(8)))  float    v8f;

#define HWPX (256 * 256)
#define HIMG 256
#define WIMG 256
#define BSZ  4
#define CIN  64
#define CATT 32

// ---- order-preserving float <-> uint encoding (for atomic min/max) ----
__device__ __forceinline__ unsigned fenc(float f) {
  unsigned u = __float_as_uint(f);
  return u ^ ((u >> 31) ? 0xFFFFFFFFu : 0x80000000u);
}
__device__ __forceinline__ float fdec(unsigned u) {
  u ^= ((u >> 31) ? 0x80000000u : 0xFFFFFFFFu);
  return __uint_as_float(u);
}

__global__ void ea_init_mm(unsigned* mm) {
  int i = threadIdx.x;
  if (i < 3) {
    mm[2 * i]     = 0xFFFFFFFFu;  // +inf encoded
    mm[2 * i + 1] = 0u;           // -inf encoded
  }
}

// global min/max of kappa[:, slice, :, :]
__global__ __launch_bounds__(256) void ea_kap_minmax(const float* __restrict__ kappa,
                                                     unsigned* __restrict__ mm) {
  const int slice = blockIdx.y;
  __shared__ unsigned smin[256], smax[256];
  unsigned lmin = 0xFFFFFFFFu, lmax = 0u;
  const int total = BSZ * HWPX;
  for (int e = blockIdx.x * blockDim.x + threadIdx.x; e < total;
       e += gridDim.x * blockDim.x) {
    int b = e / HWPX, off = e - b * HWPX;
    unsigned k = fenc(kappa[((size_t)b * 3 + slice) * HWPX + off]);
    lmin = (k < lmin) ? k : lmin;
    lmax = (k > lmax) ? k : lmax;
  }
  int tid = threadIdx.x;
  smin[tid] = lmin; smax[tid] = lmax;
  __syncthreads();
  for (int s = 128; s > 0; s >>= 1) {
    if (tid < s) {
      unsigned a = smin[tid + s]; if (a < smin[tid]) smin[tid] = a;
      unsigned b = smax[tid + s]; if (b > smax[tid]) smax[tid] = b;
    }
    __syncthreads();
  }
  if (tid == 0) {
    atomicMin(&mm[2 * slice], smin[0]);
    atomicMax(&mm[2 * slice + 1], smax[0]);
  }
}

// Assemble a 16x32 f16 WMMA fragment (A or B operand) from an LDS row of 32
// channel-pairs, per the ISA layout: halves 0-7 = K kb..kb+7, halves 8-15 =
// K kb+16..kb+23 (kb = 0 for lanes 0-15, 8 for lanes 16-31). Union with
// constant indices keeps everything in VGPRs (SROA), no scratch.
__device__ __forceinline__ v16h ea_frag(const _Float16* row, int kb2, int s) {
  const unsigned* p = (const unsigned*)row;
  union { v16h v; unsigned u[8]; } r;
#pragma unroll
  for (int j = 0; j < 4; ++j) {
    r.u[j]     = p[s * 16 + kb2 + j];
    r.u[4 + j] = p[s * 16 + kb2 + 8 + j];
  }
  return r.v;
}

// Scatter a 16x16 f32 D-matrix fragment to an LDS [row][32ch] f16 array.
__device__ __forceinline__ void ea_storeD(_Float16* dst, v8f acc, int mhi,
                                          int col, int nt) {
#pragma unroll
  for (int r = 0; r < 8; ++r)
    dst[(r + mhi) * CATT + nt * 16 + col] = (_Float16)acc[r];
}

// Fused: f16-WMMA q/k projection -> L2 normalize -> k x k windowed cosine / tau
template <int KW>
__global__ __launch_bounds__(256) void ea_affinity(const float* __restrict__ feat,
                                                   const float* __restrict__ kappa,
                                                   const float* __restrict__ qw,
                                                   const float* __restrict__ kw,
                                                   const unsigned* __restrict__ mm,
                                                   float* __restrict__ out,
                                                   int slice) {
  constexpr int P     = KW / 2;
  constexpr int H0    = 16 + 2 * P;          // halo tile edge
  constexpr int HALO2 = H0 * H0;             // halo pixels
  constexpr int KT    = (HALO2 + 15) / 16;   // 16-row M-tiles for key proj
  constexpr int PADPX = KT * 16;             // padded halo pixel count
  constexpr int KK    = KW * KW;

  extern __shared__ char lds_raw[];
  _Float16* stage = (_Float16*)lds_raw;                           // PADPX x 64 (f16)
  _Float16* wqL   = (_Float16*)(lds_raw + (size_t)PADPX * CIN * 2);
  _Float16* wkL   = wqL + CATT * CIN;
  _Float16* qs    = wkL + CATT * CIN;                             // 256 x 32 (f16)
  _Float16* ks    = qs + 256 * CATT;                              // PADPX x 32 (f16)

  const int tid = threadIdx.x;
  const int b   = blockIdx.z;
  const int y0  = blockIdx.y * 16, x0 = blockIdx.x * 16;

  // ---- stage halo of feat into LDS (f16), zero-padded OOB + tile padding ----
  const float* fb = feat + (size_t)b * CIN * HWPX;
  for (int c = 0; c < CIN; ++c) {
    const float* fc = fb + (size_t)c * HWPX;
    for (int p = tid; p < PADPX; p += 256) {
      float v = 0.f;
      if (p < HALO2) {
        int hy = p / H0, hx = p - hy * H0;
        int gy = y0 - P + hy, gx = x0 - P + hx;
        if (gy >= 0 && gy < HIMG && gx >= 0 && gx < WIMG) v = fc[gy * WIMG + gx];
      }
      stage[p * CIN + c] = (_Float16)v;
    }
  }
  for (int i = tid; i < CATT * CIN; i += 256) {
    wqL[i] = (_Float16)qw[i];
    wkL[i] = (_Float16)kw[i];
  }
  __syncthreads();

  const int lane = tid & 31;
  const int wave = tid >> 5;
  const int mrow = lane & 15;
  const int kb2  = (lane >= 16) ? 4 : 0;  // channel-pair base for this half-wave
  const int mhi  = (lane >= 16) ? 8 : 0;
  const int col  = lane & 15;

  // ---- q projection: 16 M-tiles of interior pixels ----
  {
    const _Float16* w0 = &wqL[mrow * CIN];
    const _Float16* w1 = &wqL[(16 + mrow) * CIN];
    const v16h b00 = ea_frag(w0, kb2, 0);  // kstep0, ntile0
    const v16h b10 = ea_frag(w0, kb2, 1);  // kstep1, ntile0
    const v16h b01 = ea_frag(w1, kb2, 0);  // kstep0, ntile1
    const v16h b11 = ea_frag(w1, kb2, 1);  // kstep1, ntile1
    for (int t = wave; t < 16; t += 8) {
      int ip = t * 16 + mrow;  // interior pixel 0..255
      int px = ((ip >> 4) + P) * H0 + ((ip & 15) + P);
      const _Float16* srow = &stage[px * CIN];
      v16h a0 = ea_frag(srow, kb2, 0);
      v16h a1 = ea_frag(srow, kb2, 1);
      v8f acc0 = {}, acc1 = {};
      acc0 = __builtin_amdgcn_wmma_f32_16x16x32_f16(false, a0, false, b00,
                                                    (short)0, acc0, false, false);
      acc1 = __builtin_amdgcn_wmma_f32_16x16x32_f16(false, a0, false, b01,
                                                    (short)0, acc1, false, false);
      acc0 = __builtin_amdgcn_wmma_f32_16x16x32_f16(false, a1, false, b10,
                                                    (short)0, acc0, false, false);
      acc1 = __builtin_amdgcn_wmma_f32_16x16x32_f16(false, a1, false, b11,
                                                    (short)0, acc1, false, false);
      ea_storeD(qs + (t * 16) * CATT, acc0, mhi, col, 0);
      ea_storeD(qs + (t * 16) * CATT, acc1, mhi, col, 1);
    }
  }

  // ---- key projection: KT M-tiles of halo pixels (padding rows are zero) ----
  {
    const _Float16* w0 = &wkL[mrow * CIN];
    const _Float16* w1 = &wkL[(16 + mrow) * CIN];
    const v16h b00 = ea_frag(w0, kb2, 0);
    const v16h b10 = ea_frag(w0, kb2, 1);
    const v16h b01 = ea_frag(w1, kb2, 0);
    const v16h b11 = ea_frag(w1, kb2, 1);
    for (int t = wave; t < KT; t += 8) {
      int px = t * 16 + mrow;  // halo pixel
      const _Float16* srow = &stage[px * CIN];
      v16h a0 = ea_frag(srow, kb2, 0);
      v16h a1 = ea_frag(srow, kb2, 1);
      v8f acc0 = {}, acc1 = {};
      acc0 = __builtin_amdgcn_wmma_f32_16x16x32_f16(false, a0, false, b00,
                                                    (short)0, acc0, false, false);
      acc1 = __builtin_amdgcn_wmma_f32_16x16x32_f16(false, a0, false, b01,
                                                    (short)0, acc1, false, false);
      acc0 = __builtin_amdgcn_wmma_f32_16x16x32_f16(false, a1, false, b10,
                                                    (short)0, acc0, false, false);
      acc1 = __builtin_amdgcn_wmma_f32_16x16x32_f16(false, a1, false, b11,
                                                    (short)0, acc1, false, false);
      ea_storeD(ks + (t * 16) * CATT, acc0, mhi, col, 0);
      ea_storeD(ks + (t * 16) * CATT, acc1, mhi, col, 1);
    }
  }
  __syncthreads();

  // ---- L2 normalize rows (x / max(||x||, 1e-12)); zero rows stay zero ----
  for (int r = tid; r < 256 + PADPX; r += 256) {
    _Float16* row = (r < 256) ? (qs + r * CATT) : (ks + (r - 256) * CATT);
    float s = 0.f;
#pragma unroll
    for (int c = 0; c < CATT; ++c) { float v = (float)row[c]; s += v * v; }
    float inv = 1.f / fmaxf(sqrtf(s), 1e-12f);
#pragma unroll
    for (int c = 0; c < CATT; ++c) row[c] = (_Float16)((float)row[c] * inv);
  }
  __syncthreads();

  // ---- temperature from curvature (global min/max of this slice) ----
  const float kmin  = fdec(mm[2 * slice]);
  const float kmax  = fdec(mm[2 * slice + 1]);
  const float denom = (kmax - kmin) + 1e-6f;

  const int p  = tid;          // this thread's output pixel
  const int py = p >> 4, pxx = p & 15;
  const int gy = y0 + py, gx = x0 + pxx;
  const float kap  = kappa[((size_t)b * 3 + slice) * HWPX + gy * WIMG + gx];
  float tau = 0.5f - ((kap - kmin) / denom) * (0.5f - 0.03f);
  tau = fminf(fmaxf(tau, 0.03f), 0.5f);
  const float scale = 1.f / (tau + 1e-6f);

  float qreg[CATT];
#pragma unroll
  for (int c = 0; c < CATT; ++c) qreg[c] = (float)qs[p * CATT + c];

  for (int dy = 0; dy < KW; ++dy) {
    for (int dx = 0; dx < KW; ++dx) {
      const _Float16* kr = ks + ((py + dy) * H0 + (pxx + dx)) * CATT;
      float dot = 0.f;
#pragma unroll
      for (int c = 0; c < CATT; ++c) dot += qreg[c] * (float)kr[c];
      out[(((size_t)b * KK + (dy * KW + dx)) * HWPX) + gy * WIMG + gx] = dot * scale;
    }
  }
}

static size_t ea_shmem(int KW) {
  int P = KW / 2, H0 = 16 + 2 * P;
  int KT = (H0 * H0 + 15) / 16, PADPX = KT * 16;
  return (size_t)PADPX * CIN * 2        // stage f16
       + 2 * (size_t)CATT * CIN * 2     // weights f16
       + 256 * CATT * 2                 // qn f16
       + (size_t)PADPX * CATT * 2;      // kn f16
}

extern "C" void kernel_launch(void* const* d_in, const int* in_sizes, int n_in,
                              void* d_out, int out_size, void* d_ws, size_t ws_size,
                              hipStream_t stream) {
  const float* feat  = (const float*)d_in[0];
  const float* kappa = (const float*)d_in[1];
  const float* qw    = (const float*)d_in[2];  // (3, 32, 64)
  const float* kw    = (const float*)d_in[3];  // (3, 32, 64)
  float*       out   = (float*)d_out;
  unsigned*    mm    = (unsigned*)d_ws;        // 3 x {min,max} encoded

  ea_init_mm<<<1, 64, 0, stream>>>(mm);
  ea_kap_minmax<<<dim3(256, 3), 256, 0, stream>>>(kappa, mm);

  const dim3 grid(WIMG / 16, HIMG / 16, BSZ), block(256);
  const size_t woff = (size_t)CATT * CIN;

  ea_affinity<3><<<grid, block, ea_shmem(3), stream>>>(
      feat, kappa, qw + 0 * woff, kw + 0 * woff, mm, out, 0);
  ea_affinity<5><<<grid, block, ea_shmem(5), stream>>>(
      feat, kappa, qw + 1 * woff, kw + 1 * woff, mm,
      out + (size_t)BSZ * 9 * HWPX, 1);
  ea_affinity<7><<<grid, block, ea_shmem(7), stream>>>(
      feat, kappa, qw + 2 * woff, kw + 2 * woff, mm,
      out + (size_t)BSZ * (9 + 25) * HWPX, 2);
}
